// GAT_18176301597437
// MI455X (gfx1250) — compile-verified, matrix operand order checked by reading, not run on hardware
//
#include <hip/hip_runtime.h>

#define N_NODES 4096
#define F_IN    256
#define K_HEADS 8
#define FP_DIM  64
#define D_MODEL 512           // K_HEADS * FP_DIM
#define TILE_N  64
#define TILE_M  64
#define NEG_SLOPE 0.2f
#define MASK_FILL -1e30f

typedef __attribute__((ext_vector_type(16))) _Float16 v16h;
typedef __attribute__((ext_vector_type(8)))  float    v8f;
typedef __attribute__((ext_vector_type(4)))  _Float16 v4h;

union V16H { v16h v; _Float16 h[16]; uint4 u[2]; };

__device__ __forceinline__ float eluf(float x) {
  return x > 0.f ? x : __expf(x) - 1.f;
}

// ---- CDNA5 async global->LDS (ASYNCcnt) via inline asm -----------------
// LDS destination operand is a byte offset from the wave's LDS base; the low
// 32 bits of a generic pointer to a __shared__ object are exactly that offset.
__device__ __forceinline__ unsigned lds_off(const void* p) {
  return (unsigned)(unsigned long long)p;
}
__device__ __forceinline__ void async_load_lds_b128(unsigned ldst, const void* gsrc) {
  asm volatile("global_load_async_to_lds_b128 %0, %1, off"
               :: "v"(ldst), "v"(gsrc) : "memory");
}
__device__ __forceinline__ void async_load_lds_b32(unsigned ldst, const void* gsrc) {
  asm volatile("global_load_async_to_lds_b32 %0, %1, off"
               :: "v"(ldst), "v"(gsrc) : "memory");
}
__device__ __forceinline__ void wait_async0() {
  asm volatile("s_wait_asynccnt 0x0" ::: "memory");
}

// ---------------------------------------------------------------------------
// Kernel 1: f32 -> f16 conversion (vectorized x4)
// ---------------------------------------------------------------------------
__global__ void cvt_f32_f16_kernel(const float* __restrict__ in,
                                   _Float16* __restrict__ out, int n4) {
  int i = blockIdx.x * blockDim.x + threadIdx.x;
  if (i < n4) {
    float4 f = ((const float4*)in)[i];
    v4h h = { (_Float16)f.x, (_Float16)f.y, (_Float16)f.z, (_Float16)f.w };
    ((v4h*)out)[i] = h;
  }
}

// ---------------------------------------------------------------------------
// Kernel 2: xe = x @ W^T + b  via v_wmma_f32_16x16x32_f16.
// One wave computes one 16x16 output tile; K-loop over F_IN in steps of 32.
// Stores f32 xe[n][d] (for hl/hr) and transposed f16 xet[k][f][n] (V for P@V).
// ---------------------------------------------------------------------------
__global__ __launch_bounds__(256) void xw_gemm_kernel(
    const _Float16* __restrict__ xh,   // [N][F]
    const _Float16* __restrict__ wh,   // [D][F]
    const float* __restrict__ bias,    // [D]
    float* __restrict__ xe,            // [N][D]
    _Float16* __restrict__ xet)        // [K][FP][N]
{
  const int w    = (blockIdx.x * blockDim.x + threadIdx.x) >> 5;
  const int lane = threadIdx.x & 31;
  const int rl   = lane & 15;
  const int hi   = lane >> 4;
  const int tn   = w >> 5;             // 0..255 row tiles
  const int td   = w & 31;             // 0..31  col tiles
  const int row0 = tn * 16, col0 = td * 16;

  const _Float16* ap = xh + (size_t)(row0 + rl) * F_IN + hi * 8;
  const _Float16* bp = wh + (size_t)(col0 + rl) * F_IN + hi * 8;

  v8f acc = {};
  #pragma unroll
  for (int k0 = 0; k0 < F_IN; k0 += 32) {
    V16H A, B;
    A.u[0] = *(const uint4*)(ap + k0);
    A.u[1] = *(const uint4*)(ap + k0 + 16);
    B.u[0] = *(const uint4*)(bp + k0);
    B.u[1] = *(const uint4*)(bp + k0 + 16);
    acc = __builtin_amdgcn_wmma_f32_16x16x32_f16(
        false, A.v, false, B.v, (short)0, acc, false, false);
  }

  const int col = col0 + rl;           // N index of C/D = lane&15
  const float bv = bias[col];
  const int kk = col >> 6, ff = col & 63;
  #pragma unroll
  for (int v = 0; v < 8; ++v) {
    int row = row0 + v + hi * 8;       // C/D: VGPR v -> M = v (+8 for hi lanes)
    float val = acc[v] + bv;
    xe[(size_t)row * D_MODEL + col] = val;
    xet[((size_t)kk * FP_DIM + ff) * N_NODES + row] = (_Float16)val;
  }
}

// ---------------------------------------------------------------------------
// Kernel 3: hl[k][n] = <xe[n,k,:], a_left[k,:]>, hr likewise. Wave per (k,n).
// ---------------------------------------------------------------------------
__global__ __launch_bounds__(256) void hlhr_kernel(
    const float* __restrict__ xe, const float* __restrict__ a_l,
    const float* __restrict__ a_r, float* __restrict__ hl,
    float* __restrict__ hr)
{
  const int w    = (blockIdx.x * blockDim.x + threadIdx.x) >> 5;
  const int lane = threadIdx.x & 31;
  const int n = w >> 3, k = w & 7;
  const float* xr = xe + (size_t)n * D_MODEL + k * FP_DIM;
  const float* al = a_l + k * FP_DIM;
  const float* ar = a_r + k * FP_DIM;
  float sl = 0.f, sr = 0.f;
  #pragma unroll
  for (int f = 0; f < FP_DIM; f += 32) {
    float xv = xr[f + lane];
    sl += xv * al[f + lane];
    sr += xv * ar[f + lane];
  }
  #pragma unroll
  for (int off = 16; off > 0; off >>= 1) {
    sl += __shfl_xor(sl, off, 32);
    sr += __shfl_xor(sr, off, 32);
  }
  if (lane == 0) { hl[k * N_NODES + n] = sl; hr[k * N_NODES + n] = sr; }
}

// ---------------------------------------------------------------------------
// Kernel 4: fused masked-softmax attention + P@V via WMMA (flash style).
// Block = (row-tile of 64 nodes, head). 8 waves: wave (rgrp 0..3, cgrp 0..1)
// owns rows rgrp*16..+15 and features cgrp*32..+31 (two f32 accumulators).
// Mask/hr tiles are double-buffered in LDS and staged one tile ahead with
// global_load_async_to_lds_* (ASYNCcnt), overlapping fetch with softmax+WMMA.
// ---------------------------------------------------------------------------
__global__ __launch_bounds__(256) void gat_attn_kernel(
    const float* __restrict__ hl,      // [K][N]
    const float* __restrict__ hr,      // [K][N]
    const _Float16* __restrict__ xet,  // [K][FP][N]  (V, m-contiguous)
    const unsigned char* __restrict__ mask, // [N][N] bool
    float* __restrict__ out)           // [N][K*FP]
{
  __shared__ alignas(16) unsigned char s_mask[2][TILE_N][TILE_M];
  __shared__ alignas(16) float s_hr[2][TILE_M];
  __shared__ alignas(16) float s_hl[TILE_N];

  const int k    = blockIdx.y;
  const int row0 = blockIdx.x * TILE_N;
  const int tid  = threadIdx.x;
  const int lane = tid & 31;
  const int wv   = tid >> 5;
  const int rbase = (wv >> 1) * 16;    // wave's row group
  const int fbase = (wv & 1) * 32;     // wave's feature group
  const int rl   = lane & 15;
  const int hi   = lane >> 4;

  // per-thread async staging slots
  const int mr  = tid >> 2;            // mask row within tile
  const int mc4 = (tid & 3) * 16;      // 16-byte chunk within mask row
  const unsigned char* mrow = mask + (size_t)(row0 + mr) * N_NODES + mc4;

  if (tid < TILE_N) s_hl[tid] = hl[k * N_NODES + row0 + tid];

  // preload tile 0 into buffer 0
  async_load_lds_b128(lds_off(&s_mask[0][mr][mc4]), mrow);
  if (tid < TILE_M)
    async_load_lds_b32(lds_off(&s_hr[0][tid]), hr + k * N_NODES + tid);

  __syncthreads();                     // covers s_hl
  const float hlv = s_hl[rbase + rl];  // this lane's row score-left

  const _Float16* vhead = xet + (size_t)k * FP_DIM * N_NODES;
  const _Float16* vp0 = vhead + (size_t)(fbase + rl) * N_NODES + hi * 8;
  const _Float16* vp1 = vp0 + (size_t)16 * N_NODES;

  v8f acc0 = {}, acc1 = {};
  float run_max = MASK_FILL, run_sum = 0.f;
  int buf = 0;

  for (int m0 = 0; m0 < N_NODES; m0 += TILE_M) {
    wait_async0();                     // this wave's staged tile has landed
    __syncthreads();                   // all waves landed + prev compute done

    if (m0 + TILE_M < N_NODES) {       // stage next tile into other buffer
      async_load_lds_b128(lds_off(&s_mask[buf ^ 1][mr][mc4]),
                          mrow + m0 + TILE_M);
      if (tid < TILE_M)
        async_load_lds_b32(lds_off(&s_hr[buf ^ 1][tid]),
                           hr + k * N_NODES + m0 + TILE_M + tid);
      __builtin_prefetch(vp0 + m0 + TILE_M, 0, 1);
      __builtin_prefetch(vp1 + m0 + TILE_M, 0, 1);
    }

    // ---- scores for this lane's 32 (row, m) slots (A-operand striping) ----
    float earr[32];
    #pragma unroll
    for (int c = 0; c < 2; ++c) {
      #pragma unroll
      for (int g = 0; g < 2; ++g) {
        const int mb = c * 32 + hi * 8 + g * 16;
        float hrv[8];
        *(float4*)&hrv[0] = *(const float4*)&s_hr[buf][mb];
        *(float4*)&hrv[4] = *(const float4*)&s_hr[buf][mb + 4];
        uint2 mv = *(const uint2*)&s_mask[buf][rbase + rl][mb];
        #pragma unroll
        for (int j = 0; j < 8; ++j) {
          float ev = hlv + hrv[j];
          ev = ev > 0.f ? ev : NEG_SLOPE * ev;                 // LeakyReLU
          unsigned byte = (j < 4) ? (mv.x >> (j * 8)) : (mv.y >> ((j - 4) * 8));
          earr[c * 16 + g * 8 + j] = (byte & 0xffu) ? ev : MASK_FILL;
        }
      }
    }

    // ---- online softmax: tile max over lane pair (l, l+16) ----
    float tmax = MASK_FILL;
    #pragma unroll
    for (int i = 0; i < 32; ++i) tmax = fmaxf(tmax, earr[i]);
    tmax = fmaxf(tmax, __shfl_xor(tmax, 16, 32));
    const float nmax  = fmaxf(run_max, tmax);
    const float alpha = __expf(run_max - nmax);
    run_max = nmax;

    V16H P[2];
    float tsum = 0.f;
    #pragma unroll
    for (int c = 0; c < 2; ++c)
      #pragma unroll
      for (int i = 0; i < 16; ++i) {
        float pv = __expf(earr[c * 16 + i] - nmax);
        tsum += pv;
        P[c].h[i] = (_Float16)pv;
      }
    tsum += __shfl_xor(tsum, 16, 32);
    run_sum = run_sum * alpha + tsum;

    // rescale accumulators: VGPR v holds row v+8*hi; alpha[row] lives in lane=row
    #pragma unroll
    for (int v = 0; v < 8; ++v) {
      float av = __shfl(alpha, v + hi * 8, 32);
      acc0[v] *= av;
      acc1[v] *= av;
    }

    // ---- P @ V : 2 K-chunks x 2 feature tiles of wmma_f32_16x16x32_f16 ----
    #pragma unroll
    for (int c = 0; c < 2; ++c) {
      V16H B0, B1;
      const _Float16* b0 = vp0 + m0 + c * 32;   // col f = fbase+rl, m-contig
      const _Float16* b1 = vp1 + m0 + c * 32;   // col f = fbase+16+rl
      B0.u[0] = *(const uint4*)(b0);
      B0.u[1] = *(const uint4*)(b0 + 16);
      B1.u[0] = *(const uint4*)(b1);
      B1.u[1] = *(const uint4*)(b1 + 16);
      acc0 = __builtin_amdgcn_wmma_f32_16x16x32_f16(
          false, P[c].v, false, B0.v, (short)0, acc0, false, false);
      acc1 = __builtin_amdgcn_wmma_f32_16x16x32_f16(
          false, P[c].v, false, B1.v, (short)0, acc1, false, false);
    }

    buf ^= 1;
  }

  // ---- epilogue: normalize, elu(elu(.)), store ----
  const float invl = 1.0f / run_sum;   // denominator for row rl (dup in halves)
  #pragma unroll
  for (int v = 0; v < 8; ++v) {
    float iv = __shfl(invl, v + hi * 8, 32);
    int row = row0 + rbase + v + hi * 8;
    float x0 = eluf(eluf(acc0[v] * iv));
    float x1 = eluf(eluf(acc1[v] * iv));
    float* orow = out + (size_t)row * D_MODEL + k * FP_DIM + fbase;
    orow[rl]      = x0;
    orow[16 + rl] = x1;
  }
}

// ---------------------------------------------------------------------------
// Kernel 5: mask passthrough (tuple output #2), bool -> float
// ---------------------------------------------------------------------------
__global__ void mask_out_kernel(const unsigned char* __restrict__ m,
                                float* __restrict__ out, int n4) {
  int i = blockIdx.x * blockDim.x + threadIdx.x;
  if (i < n4) {
    uchar4 v = *(const uchar4*)(m + (size_t)i * 4);
    float4 f = { v.x ? 1.f : 0.f, v.y ? 1.f : 0.f,
                 v.z ? 1.f : 0.f, v.w ? 1.f : 0.f };
    *(float4*)(out + (size_t)i * 4) = f;
  }
}

// ---------------------------------------------------------------------------
extern "C" void kernel_launch(void* const* d_in, const int* in_sizes, int n_in,
                              void* d_out, int out_size, void* d_ws, size_t ws_size,
                              hipStream_t stream) {
  (void)in_sizes; (void)n_in; (void)out_size; (void)ws_size;
  const float* x   = (const float*)d_in[0];
  const float* W_w = (const float*)d_in[1];
  const float* W_b = (const float*)d_in[2];
  const float* a_l = (const float*)d_in[3];
  const float* a_r = (const float*)d_in[4];
  const unsigned char* mask = (const unsigned char*)d_in[5];
  float* out = (float*)d_out;

  char* ws = (char*)d_ws;
  size_t off = 0;
  _Float16* x_h = (_Float16*)(ws + off); off += (size_t)N_NODES * F_IN * 2;   // 2 MB
  _Float16* w_h = (_Float16*)(ws + off); off += (size_t)D_MODEL * F_IN * 2;   // 256 KB
  float*    xe  = (float*)   (ws + off); off += (size_t)N_NODES * D_MODEL * 4;// 8 MB
  _Float16* xet = (_Float16*)(ws + off); off += (size_t)N_NODES * D_MODEL * 2;// 4 MB
  float*    hl  = (float*)   (ws + off); off += (size_t)K_HEADS * N_NODES * 4;
  float*    hr  = (float*)   (ws + off); off += (size_t)K_HEADS * N_NODES * 4;

  cvt_f32_f16_kernel<<<(N_NODES * F_IN / 4 + 255) / 256, 256, 0, stream>>>(
      x, x_h, N_NODES * F_IN / 4);
  cvt_f32_f16_kernel<<<(D_MODEL * F_IN / 4 + 255) / 256, 256, 0, stream>>>(
      W_w, w_h, D_MODEL * F_IN / 4);

  // (N/16)*(D/16) = 8192 waves -> 1024 blocks of 8 waves
  xw_gemm_kernel<<<1024, 256, 0, stream>>>(x_h, w_h, W_b, xe, xet);

  // N*K = 32768 waves -> 4096 blocks
  hlhr_kernel<<<4096, 256, 0, stream>>>(xe, a_l, a_r, hl, hr);

  gat_attn_kernel<<<dim3(N_NODES / TILE_N, K_HEADS), 256, 0, stream>>>(
      hl, hr, xet, mask, out);

  mask_out_kernel<<<(N_NODES * N_NODES / 4 + 255) / 256, 256, 0, stream>>>(
      mask, out + (size_t)N_NODES * D_MODEL, N_NODES * N_NODES / 4);
}
